// NequIP_dpm_cond_72894184948209
// MI455X (gfx1250) — compile-verified, matrix operand order checked by reading, not run on hardware
//
#include <hip/hip_runtime.h>

typedef __attribute__((ext_vector_type(16))) _Float16 v16h;
typedef __attribute__((ext_vector_type(8)))  _Float16 v8h;
typedef __attribute__((ext_vector_type(8)))  float    v8f;
typedef __attribute__((ext_vector_type(4)))  float    v4f;
typedef __attribute__((ext_vector_type(4)))  unsigned int u32x4;
typedef __attribute__((ext_vector_type(8)))  int      i32x8;
typedef __attribute__((ext_vector_type(4)))  int      i32x4;

#define NNODES 30000
#define NEDGES 480000
#define PV (NNODES*32)     // vector-plane stride (ld=32)
#define PA (NNODES*128)    // a1 plane stride

#define WMMA_F16(a, b, c) __builtin_amdgcn_wmma_f32_16x16x32_f16(false, (a), false, (b), (short)0, (c), false, false)

// ---------------- fragment helpers ----------------
// CDNA5 16-bit A/B frag: lane holds row (A: m=lane&15 / B^T: n=lane&15),
// K slots = {kb..kb+7, kb+16..kb+23}, kb = (lane&16)?8:0.
__device__ __forceinline__ v16h frag_h(const _Float16* row, int lane) {
    int kb = (lane & 16) ? 8 : 0;
    v8h lo = *(const v8h*)(row + kb);
    v8h hi = *(const v8h*)(row + kb + 16);
    return __builtin_shufflevector(lo, hi, 0, 1, 2, 3, 4, 5, 6, 7, 8, 9, 10, 11, 12, 13, 14, 15);
}
__device__ __forceinline__ v16h frag_f32(const float* row, int lane) {
    int kb = (lane & 16) ? 8 : 0;
    v4f a0 = *(const v4f*)(row + kb);
    v4f a1 = *(const v4f*)(row + kb + 4);
    v4f b0 = *(const v4f*)(row + kb + 16);
    v4f b1 = *(const v4f*)(row + kb + 20);
    v16h r;
#pragma unroll
    for (int i = 0; i < 4; i++) {
        r[i]      = (_Float16)a0[i];
        r[i + 4]  = (_Float16)a1[i];
        r[i + 8]  = (_Float16)b0[i];
        r[i + 12] = (_Float16)b1[i];
    }
    return r;
}

// ---------------- Tensor Data Mover: 2D f32 tile -> LDS (D# per ISA 8.3/8.4) ----------------
__device__ __forceinline__ unsigned lds_off(const void* p) {
    return (unsigned)(unsigned long long)(p);
}
__device__ __forceinline__ void tdm_load_2d_f32(unsigned lds_byte, const void* gaddr,
                                                unsigned tensor_d0, unsigned tensor_d1,
                                                unsigned tile_d0, unsigned tile_d1,
                                                unsigned stride_elems) {
    unsigned long long ga = (unsigned long long)gaddr;
    u32x4 g0;
    g0[0] = 1u;                                        // count=1
    g0[1] = lds_byte;                                  // lds_addr
    g0[2] = (unsigned)(ga & 0xffffffffu);              // global_addr[31:0]
    g0[3] = (unsigned)((ga >> 32) & 0x1ffffffu) | (2u << 30);  // [56:32], type=2
    i32x8 g1;
    g1[0] = (int)(2u << 16);                           // data_size=4B
    g1[1] = (int)((tensor_d0 & 0xffffu) << 16);
    g1[2] = (int)((tensor_d0 >> 16) | ((tensor_d1 & 0xffffu) << 16));
    g1[3] = (int)((tensor_d1 >> 16) | (tile_d0 << 16));
    g1[4] = (int)(tile_d1 & 0xffffu);
    g1[5] = (int)stride_elems;
    g1[6] = 0;
    g1[7] = 0;
    i32x4 z4 = {0, 0, 0, 0};
#if defined(__clang_major__) && __clang_major__ >= 23
    i32x8 z8 = {0, 0, 0, 0, 0, 0, 0, 0};
    __builtin_amdgcn_tensor_load_to_lds(g0, g1, z4, z4, z8, 0);
#else
    __builtin_amdgcn_tensor_load_to_lds(g0, g1, z4, z4, 0);
#endif
}

// ---------------- weight prep: out[n*Kp+k] = f16(op(in)[k][n]), zero-padded ----------------
__global__ void prep_bT(const float* __restrict__ in, _Float16* __restrict__ out,
                        int K, int Nn, int Kp, int Np, int ldin, int srcT) {
    int idx = blockIdx.x * blockDim.x + threadIdx.x;
    if (idx >= Kp * Np) return;
    int nn = idx / Kp, k = idx - nn * Kp;
    float v = 0.f;
    if (k < K && nn < Nn) v = srcT ? in[nn * ldin + k] : in[k * ldin + nn];
    out[idx] = (_Float16)v;
}

// ---------------- time embedding (tiny) ----------------
__global__ void temb_kernel(const float* __restrict__ t, const float* __restrict__ gfp,
                            const float* __restrict__ w1, const float* __restrict__ b1,
                            const float* __restrict__ w2, const float* __restrict__ b2,
                            float* __restrict__ temb) {
    __shared__ float e[16 * 128];
    __shared__ float h[16 * 128];
    int tid = threadIdx.x;
    for (int idx = tid; idx < 2048; idx += 128) {
        int g = idx >> 7, i = idx & 127;
        float pr = t[g] * gfp[i & 63] * 6.2831853071795864f;
        e[idx] = (i < 64) ? __sinf(pr) : __cosf(pr);
    }
    __syncthreads();
    for (int idx = tid; idx < 2048; idx += 128) {
        int g = idx >> 7, j = idx & 127;
        float acc = b1[j];
        for (int k = 0; k < 128; k++) acc += e[g * 128 + k] * w1[k * 128 + j];
        h[idx] = acc / (1.f + __expf(-acc));
    }
    __syncthreads();
    for (int idx = tid; idx < 1024; idx += 128) {
        int g = idx >> 6, c = idx & 63;
        float acc = b2[c];
        for (int k = 0; k < 128; k++) acc += h[g * 128 + k] * w2[k * 64 + c];
        temb[g * 64 + c] = acc;
    }
}

// ---------------- node init ----------------
__global__ void init_kernel(const float* __restrict__ x, const int* __restrict__ atm,
                            const int* __restrict__ batch, const float* __restrict__ embx,
                            const float* __restrict__ embz, const float* __restrict__ temb,
                            float* __restrict__ z, float* __restrict__ s, float* __restrict__ v) {
    int idx = blockIdx.x * blockDim.x + threadIdx.x;
    if (idx >= NNODES * 64) return;
    int n = idx >> 6, c = idx & 63;
    int a = atm[n];
    z[idx] = embz[a * 64 + c] + temb[batch[n] * 64 + c];
    if (c < 16) s[n * 64 + c] = embx[a * 16 + c];
    if (c < 3)  v[c * PV + n * 32 + 0] = x[n * 3 + c];
}

// ---------------- GEMM: C (+)= A[MxK] * B, B pre-transposed f16 [n][Kp] ----------------
// Per wave: 16x32 C strip (2 accumulators). A tile double-buffered via TDM.
__global__ void __launch_bounds__(128) wmma_gemm_f32(const float* __restrict__ A,
                                                     const _Float16* __restrict__ BT,
                                                     float* __restrict__ C,
                                                     int M, int N, int K, int lda, int ldc,
                                                     int ldbt, int accum) {
    __shared__ float As[4][2][16 * 32];
    int lane = threadIdx.x & 31, wv = threadIdx.x >> 5;
    int ntn2 = (N + 31) >> 5;
    int tiles = ((M + 15) >> 4) * ntn2;
    int tile = blockIdx.x * 4 + wv;
    int active = tile < tiles;
    int tm = 0, tn2 = 0;
    if (active) { tm = tile / ntn2; tn2 = tile - tm * ntn2; }
    int n = lane & 15;
    const _Float16* B0 = BT + (size_t)(tn2 * 32 + n) * ldbt;
    const _Float16* B1 = BT + (size_t)(tn2 * 32 + 16 + n) * ldbt;
    const float* Abase = A + (size_t)(tm * 16) * lda;
    unsigned l0 = lds_off(As[wv][0]), l1 = lds_off(As[wv][1]);
    v8f acc0 = {}, acc1 = {};
    int nk = (K + 31) >> 5;
    tdm_load_2d_f32(l0, Abase, (unsigned)K, (unsigned)(M - tm * 16), 32u, 16u, (unsigned)lda);
    for (int t = 0; t < nk; t++) {
        int k0 = t * 32;
        const float* curp = (t & 1) ? As[wv][1] : As[wv][0];
        if (t + 1 < nk) {
            tdm_load_2d_f32((t & 1) ? l0 : l1, Abase + k0 + 32,
                            (unsigned)(K - k0 - 32), (unsigned)(M - tm * 16), 32u, 16u, (unsigned)lda);
            __builtin_amdgcn_s_wait_tensorcnt(1);
        } else {
            __builtin_amdgcn_s_wait_tensorcnt(0);
        }
        __syncthreads();
        v16h a  = frag_f32(curp + n * 32, lane);
        v16h b0 = frag_h(B0 + k0, lane);
        v16h b1 = frag_h(B1 + k0, lane);
        acc0 = WMMA_F16(a, b0, acc0);
        acc1 = WMMA_F16(a, b1, acc1);
    }
    if (active) {
        int mb = (lane & 16) ? 8 : 0;
#pragma unroll
        for (int r = 0; r < 8; r++) {
            int gm = tm * 16 + r + mb;
            if (gm < M) {
                int gn0 = tn2 * 32 + n, gn1 = gn0 + 16;
                if (gn0 < N) {
                    float o = acc0[r];
                    if (accum) o += C[(size_t)gm * ldc + gn0];
                    C[(size_t)gm * ldc + gn0] = o;
                }
                if (gn1 < N) {
                    float o = acc1[r];
                    if (accum) o += C[(size_t)gm * ldc + gn1];
                    C[(size_t)gm * ldc + gn1] = o;
                }
            }
        }
    }
}

// ---------------- bilinear: C[n,w] += sum_{u,v} X[n,u]*Z[n,v]*WtT[w][u*64+v] ----------------
__global__ void __launch_bounds__(128) wmma_bilinear(const float* __restrict__ X, int ldx, int mx,
                                                     const float* __restrict__ Z,
                                                     const _Float16* __restrict__ WtT, int nw,
                                                     float* __restrict__ C, int ldc, int M) {
    __shared__ float shX[4][16 * 64];
    __shared__ float shZ[4][16 * 64];
    int lane = threadIdx.x & 31, wv = threadIdx.x >> 5;
    int ntn = (nw + 15) >> 4;
    int tiles = ((M + 15) >> 4) * ntn;
    int tile = blockIdx.x * 4 + wv;
    int active = tile < tiles;
    int tm = 0, tn = 0;
    if (active) { tm = tile / ntn; tn = tile - tm * ntn; }
    float* Xs = shX[wv];
    float* Zs = shZ[wv];
    for (int idx = lane; idx < 16 * 64; idx += 32) {
        int r = idx >> 6, c = idx & 63;
        int gr = tm * 16 + r;
        int grc = (gr < M) ? gr : M - 1;
        float xv = (c < mx) ? X[(size_t)grc * ldx + c] : 0.f;
        Xs[idx] = (gr < M) ? xv : 0.f;
        Zs[idx] = (gr < M) ? Z[(size_t)grc * 64 + c] : 0.f;
    }
    __syncthreads();
    v8f acc = {};
    int Ktot = mx * 64;
    int m = lane & 15, n = lane & 15;
    const _Float16* Brow = WtT + (size_t)(tn * 16 + n) * Ktot;
    int kb = (lane & 16) ? 8 : 0;
    for (int k0 = 0; k0 < Ktot; k0 += 32) {
        int ka = k0 + kb, kc = ka + 16;
        float xu0 = Xs[m * 64 + (ka >> 6)];
        float xu1 = Xs[m * 64 + (kc >> 6)];
        v4f z00 = *(const v4f*)(Zs + m * 64 + (ka & 63));
        v4f z01 = *(const v4f*)(Zs + m * 64 + (ka & 63) + 4);
        v4f z10 = *(const v4f*)(Zs + m * 64 + (kc & 63));
        v4f z11 = *(const v4f*)(Zs + m * 64 + (kc & 63) + 4);
        v16h a;
#pragma unroll
        for (int i = 0; i < 4; i++) {
            a[i]      = (_Float16)(xu0 * z00[i]);
            a[i + 4]  = (_Float16)(xu0 * z01[i]);
            a[i + 8]  = (_Float16)(xu1 * z10[i]);
            a[i + 12] = (_Float16)(xu1 * z11[i]);
        }
        v16h b = frag_h(Brow + k0, lane);
        acc = WMMA_F16(a, b, acc);
    }
    if (active) {
        int mb = (lane & 16) ? 8 : 0;
        int gn = tn * 16 + (lane & 15);
        if (gn < nw) {
#pragma unroll
            for (int r = 0; r < 8; r++) {
                int gm = tm * 16 + r + mb;
                if (gm < M) C[(size_t)gm * ldc + gn] += acc[r];
            }
        }
    }
}

// ---------------- fused edge kernel ----------------
struct EdgeSh {
    _Float16 rbf[16 * 32];
    _Float16 h[16 * 64];
    float    wt[16 * 16];
    float    y1[16 * 3];
    float    r[16];
    int      src[16];
    int      dst[16];
    int      valid[16];
};

__global__ void __launch_bounds__(128) edge_kernel(const float* __restrict__ evec,
                                                   const int* __restrict__ srcI, const int* __restrict__ dstI,
                                                   const float* __restrict__ s1, const float* __restrict__ v1,
                                                   const float* __restrict__ fw1, const float* __restrict__ fb1,
                                                   const float* __restrict__ fw2,
                                                   float* __restrict__ a0, float* __restrict__ a1,
                                                   int m0, int m1, int nw, int nE, int a0w, int a1w) {
    __shared__ EdgeSh sh[4];
    __shared__ _Float16 w1t[64 * 32];
    __shared__ _Float16 w2t[224 * 64];
    int lane = threadIdx.x & 31, wv = threadIdx.x >> 5;
    int tid = threadIdx.x;
    EdgeSh& W = sh[wv];
    int ebase = (blockIdx.x * 4 + wv) * 16;

    for (int idx = tid; idx < 64 * 32; idx += 128) {
        int nn = idx >> 5, k = idx & 31;
        w1t[idx] = (_Float16)fw1[k * 64 + nn];
    }
    for (int idx = tid; idx < 224 * 64; idx += 128) {
        int nn = idx >> 6, k = idx & 63;
        int nnc = (nn < nw) ? nn : 0;
        float v = fw2[(size_t)k * nw + nnc];
        w2t[idx] = (_Float16)((nn < nw) ? v : 0.f);
    }

    if (lane < 16) {
        int e = ebase + lane;
        int val = e < nE;
        int ec = val ? e : 0;
        int s = srcI[ec], d = dstI[ec];
        float vx = evec[ec * 3], vy = evec[ec * 3 + 1], vz = evec[ec * 3 + 2];
        float r = sqrtf(vx * vx + vy * vy + vz * vz);
        if (!(r > 0.f)) r = 1.f;
        const float SQ3 = 1.7320508075688772f;
        W.y1[lane * 3 + 0] = SQ3 * vx / r;
        W.y1[lane * 3 + 1] = SQ3 * vy / r;
        W.y1[lane * 3 + 2] = SQ3 * vz / r;
        W.r[lane] = r; W.src[lane] = s; W.dst[lane] = d; W.valid[lane] = val;
        __builtin_prefetch(&s1[s * 64], 0, 1);
    }
    __syncthreads();

    for (int idx = lane; idx < 512; idx += 32) {
        int e = idx >> 5, j = idx & 31;
        float r = W.r[e];
        const float step = 5.0f / 33.0f;
        float d = (r - step * (float)(j + 1)) / step;
        float u1 = d + 1.f, u2 = 1.f - d;
        float f1 = (u1 > 0.f) ? __expf(-1.f / u1) : 0.f;
        float f2 = (u2 > 0.f) ? __expf(-1.f / u2) : 0.f;
        W.rbf[idx] = (_Float16)(8.433800233f * f1 * f2 * 5.656854249f);
    }
    __syncthreads();

    int n = lane & 15, mb = (lane & 16) ? 8 : 0;
    v16h arb = frag_h(W.rbf + n * 32, lane);
#pragma unroll
    for (int nt = 0; nt < 4; nt++) {
        v16h b = frag_h(w1t + (nt * 16 + n) * 32, lane);
        v8f c = {};
        c = WMMA_F16(arb, b, c);
        float bias = fb1[nt * 16 + n];
#pragma unroll
        for (int r8 = 0; r8 < 8; r8++) {
            float xv = c[r8] + bias;
            W.h[(r8 + mb) * 64 + nt * 16 + n] = (_Float16)(xv / (1.f + __expf(-xv)));
        }
    }
    __syncthreads();

    v16h ah0 = frag_h(W.h + n * 64, lane);
    v16h ah1 = frag_h(W.h + n * 64 + 32, lane);
    int ntw = (nw + 15) >> 4;
    const float inv = 0.25f;

    for (int t = 0; t < ntw; t++) {
        v8f c = {};
        c = WMMA_F16(ah0, frag_h(w2t + (t * 16 + n) * 64, lane), c);
        c = WMMA_F16(ah1, frag_h(w2t + (t * 16 + n) * 64 + 32, lane), c);
#pragma unroll
        for (int r8 = 0; r8 < 8; r8++) W.wt[(r8 + mb) * 16 + n] = c[r8];
        __syncthreads();

#pragma unroll
        for (int q = 0; q < 8; q++) {
            int idx = q * 32 + lane;
            int e = idx >> 4, cc = idx & 15;
            int ch = t * 16 + cc;
            if (ch < nw && W.valid[e]) {
                float wv_ = W.wt[e * 16 + cc] * inv;
                int sN = W.src[e], dN = W.dst[e];
                float y0 = W.y1[e * 3], yy1 = W.y1[e * 3 + 1], y2 = W.y1[e * 3 + 2];
                if (ch < m0) {
                    atomicAdd(&a0[dN * a0w + ch], wv_ * s1[sN * 64 + ch]);
                } else if (ch < m0 + m1) {
                    int u = ch - m0;
                    float d0 = v1[0 * PV + sN * 32 + u], d1 = v1[1 * PV + sN * 32 + u], d2 = v1[2 * PV + sN * 32 + u];
                    atomicAdd(&a0[dN * a0w + m0 + u], wv_ * (d0 * y0 + d1 * yy1 + d2 * y2));
                } else if (ch < 2 * m0 + m1) {
                    int u = ch - (m0 + m1);
                    float ss = wv_ * s1[sN * 64 + u];
                    atomicAdd(&a1[0 * PA + dN * a1w + u], ss * y0);
                    atomicAdd(&a1[1 * PA + dN * a1w + u], ss * yy1);
                    atomicAdd(&a1[2 * PA + dN * a1w + u], ss * y2);
                } else if (ch < 2 * m0 + 2 * m1) {
                    int u = ch - (2 * m0 + m1);
                    atomicAdd(&a1[0 * PA + dN * a1w + m0 + u], wv_ * v1[0 * PV + sN * 32 + u]);
                    atomicAdd(&a1[1 * PA + dN * a1w + m0 + u], wv_ * v1[1 * PV + sN * 32 + u]);
                    atomicAdd(&a1[2 * PA + dN * a1w + m0 + u], wv_ * v1[2 * PV + sN * 32 + u]);
                } else {
                    int u = ch - (2 * m0 + 2 * m1);
                    float d0 = v1[0 * PV + sN * 32 + u], d1 = v1[1 * PV + sN * 32 + u], d2 = v1[2 * PV + sN * 32 + u];
                    atomicAdd(&a1[0 * PA + dN * a1w + m0 + m1 + u], wv_ * (d1 * y2 - d2 * yy1));
                    atomicAdd(&a1[1 * PA + dN * a1w + m0 + m1 + u], wv_ * (d2 * y0 - d0 * y2));
                    atomicAdd(&a1[2 * PA + dN * a1w + m0 + m1 + u], wv_ * (d0 * yy1 - d1 * y0));
                }
            }
        }
        __syncthreads();
    }
}

// ---------------- gated activation ----------------
__global__ void act_kernel(const float* __restrict__ hs, const float* __restrict__ hv,
                           float* __restrict__ sout, float* __restrict__ vout) {
    int idx = blockIdx.x * blockDim.x + threadIdx.x;
    if (idx >= NNODES * 96) return;
    int nn = idx / 96, j = idx - nn * 96;
    if (j < 64) {
        float xv = hs[nn * 96 + j];
        sout[nn * 64 + j] = xv / (1.f + __expf(-xv));
    } else {
        int u = j - 64;
        float g = 1.f / (1.f + __expf(-hs[nn * 96 + 64 + u]));
#pragma unroll
        for (int d = 0; d < 3; d++)
            vout[d * PV + nn * 32 + u] = g * hv[d * (NNODES * 32) + nn * 32 + u];
    }
}

// ---------------- output ----------------
__global__ void out_kernel(const float* __restrict__ v, const float* __restrict__ q,
                           float* __restrict__ out) {
    int idx = blockIdx.x * blockDim.x + threadIdx.x;
    if (idx >= NNODES * 3) return;
    int nn = idx / 3, d = idx - nn * 3;
    float acc = 0.f;
#pragma unroll
    for (int u = 0; u < 32; u++) acc += v[d * PV + nn * 32 + u] * q[nn * 32 + u];
    out[nn * 3 + d] = acc;
}

// ======================= host =======================
extern "C" void kernel_launch(void* const* d_in, const int* in_sizes, int n_in,
                              void* d_out, int out_size, void* d_ws, size_t ws_size,
                              hipStream_t stream) {
    (void)in_sizes; (void)n_in; (void)out_size; (void)ws_size;
    const float* x      = (const float*)d_in[0];
    const int*   x_atm  = (const int*)d_in[1];
    const int*   eidx   = (const int*)d_in[2];
    const float* evec   = (const float*)d_in[3];
    const int*   batch  = (const int*)d_in[4];
    const float* t      = (const float*)d_in[5];
    const float* embx   = (const float*)d_in[6];
    const float* embz   = (const float*)d_in[7];
    const float* gfp    = (const float*)d_in[8];
    const float* tw1    = (const float*)d_in[9];
    const float* tb1    = (const float*)d_in[10];
    const float* tw2    = (const float*)d_in[11];
    const float* tb2    = (const float*)d_in[12];
    const float* outw   = (const float*)d_in[13];

    float* W = (float*)d_ws;
    size_t off = 0;
    auto alloc = [&](size_t nfl) { off = (off + 7) & ~(size_t)7; float* p = W + off; off += nfl; return p; };
    float* temb = alloc(16 * 64);
    float* z    = alloc((size_t)NNODES * 64);
    float* sA   = alloc((size_t)NNODES * 64);
    float* sB   = alloc((size_t)NNODES * 64);
    float* vA   = alloc((size_t)3 * NNODES * 32);
    float* vB   = alloc((size_t)3 * NNODES * 32);
    float* s1   = alloc((size_t)NNODES * 64);
    float* v1   = alloc((size_t)3 * NNODES * 32);
    float* a0   = alloc((size_t)NNODES * 96);
    float* a1   = alloc((size_t)3 * NNODES * 128);
    float* hs   = alloc((size_t)NNODES * 96);
    float* hv   = alloc((size_t)3 * NNODES * 32);
    float* q    = alloc((size_t)NNODES * 32);
    // f16 pre-transposed weights (counted in float units, 2 f16 per float)
    _Float16* bT1s  = (_Float16*)alloc(64 * 64 / 2);
    _Float16* bT1v  = (_Float16*)alloc(32 * 32 / 2);
    _Float16* bT2s  = (_Float16*)alloc(96 * 96 / 2 + 8);
    _Float16* bT2v  = (_Float16*)alloc(128 * 32 / 2);
    _Float16* bTout = (_Float16*)alloc(64 * 32 / 2);
    _Float16* scsT  = (_Float16*)alloc(4096 * 96 / 2);
    _Float16* scvT  = (_Float16*)alloc(2048 * 32 / 2);

    auto ceil32 = [](int v) { return (v + 31) & ~31; };
    auto prep = [&](const float* in, _Float16* o, int K, int Nn, int Kp, int Np, int ldin, int sT) {
        int tot = Kp * Np;
        prep_bT<<<(tot + 255) / 256, 256, 0, stream>>>(in, o, K, Nn, Kp, Np, ldin, sT);
    };
    auto gemm = [&](const float* A, const _Float16* BT, float* C, int M, int N, int K,
                    int lda, int ldc, int ldbt, int acc) {
        int tiles = ((M + 15) / 16) * ((N + 31) / 32);
        int blocks = (tiles + 3) / 4;
        wmma_gemm_f32<<<blocks, 128, 0, stream>>>(A, BT, C, M, N, K, lda, ldc, ldbt, acc);
    };
    auto bilin = [&](const float* X, int ldx, int mx, const _Float16* WtT, int nwout, float* C, int ldc) {
        int tiles = ((NNODES + 15) / 16) * ((nwout + 15) / 16);
        int blocks = (tiles + 3) / 4;
        wmma_bilinear<<<blocks, 128, 0, stream>>>(X, ldx, mx, z, WtT, nwout, C, ldc, NNODES);
    };

    temb_kernel<<<1, 128, 0, stream>>>(t, gfp, tw1, tb1, tw2, tb2, temb);
    init_kernel<<<(NNODES * 64 + 255) / 256, 256, 0, stream>>>(x, x_atm, batch, embx, embz, temb, z, sA, vA);
    prep(outw, bTout, 64, 32, 64, 32, 64, 1);   // q = z @ out_w^T : B[k][n]=out_w[n][k]

    for (int L = 0; L < 3; L++) {
        int m0 = L ? 64 : 16, m1 = L ? 32 : 1;
        int nw = 2 * m0 + 3 * m1;
        int a0w = m0 + m1, a1w = m0 + 2 * m1;
        const float* sin_ = (L & 1) ? sB : sA;
        float*       sout = (L & 1) ? sA : sB;
        const float* vin  = (L & 1) ? vB : vA;
        float*       vout = (L & 1) ? vA : vB;
        int base = 14 + 9 * L;
        const float* lin1_s = (const float*)d_in[base + 0];
        const float* lin1_v = (const float*)d_in[base + 1];
        const float* fw1    = (const float*)d_in[base + 2];
        const float* fb1    = (const float*)d_in[base + 3];
        const float* fw2    = (const float*)d_in[base + 4];
        const float* lin2_s = (const float*)d_in[base + 5];
        const float* lin2_v = (const float*)d_in[base + 6];
        const float* sc_s   = (const float*)d_in[base + 7];
        const float* sc_v   = (const float*)d_in[base + 8];

        int k1s = ceil32(m0), k1v = ceil32(m1);
        int k2s = ceil32(a0w), k2v = ceil32(a1w);
        prep(lin1_s, bT1s, m0, m0, k1s, ceil32(m0), m0, 0);
        prep(lin1_v, bT1v, m1, m1, k1v, ceil32(m1), m1, 0);
        prep(lin2_s, bT2s, a0w, 96, k2s, 96, 96, 0);
        prep(lin2_v, bT2v, a1w, 32, k2v, 32, 32, 0);
        prep(sc_s, scsT, m0 * 64, 96, m0 * 64, 96, 96, 0);
        prep(sc_v, scvT, m1 * 64, 32, m1 * 64, 32, 32, 0);

        gemm(sin_, bT1s, s1, NNODES, m0, m0, 64, 64, k1s, 0);
        for (int d = 0; d < 3; d++)
            gemm(vin + (size_t)d * PV, bT1v, v1 + (size_t)d * PV, NNODES, m1, m1, 32, 32, k1v, 0);

        hipMemsetAsync(a0, 0, (size_t)NNODES * 96 * sizeof(float), stream);
        hipMemsetAsync(a1, 0, (size_t)3 * NNODES * 128 * sizeof(float), stream);

        edge_kernel<<<(NEDGES + 63) / 64, 128, 0, stream>>>(evec, eidx, eidx + NEDGES, s1, v1,
                                                            fw1, fb1, fw2, a0, a1,
                                                            m0, m1, nw, NEDGES, a0w, a1w);

        gemm(a0, bT2s, hs, NNODES, 96, a0w, a0w, 96, k2s, 0);
        bilin(sin_, 64, m0, scsT, 96, hs, 96);
        for (int d = 0; d < 3; d++) {
            gemm(a1 + (size_t)d * PA, bT2v, hv + (size_t)d * NNODES * 32, NNODES, 32, a1w, a1w, 32, k2v, 0);
            bilin(vin + (size_t)d * PV, 32, m1, scvT, 32, hv + (size_t)d * NNODES * 32, 32);
        }
        act_kernel<<<(NNODES * 96 + 255) / 256, 256, 0, stream>>>(hs, hv, sout, vout);
    }

    gemm(z, bTout, q, NNODES, 32, 64, 64, 32, 64, 0);
    out_kernel<<<(NNODES * 3 + 255) / 256, 256, 0, stream>>>(vB, q, (float*)d_out);
}
